// FixedProductionSplatFlowAttention_43224550867608
// MI455X (gfx1250) — compile-verified, matrix operand order checked by reading, not run on hardware
//
#include <hip/hip_runtime.h>
#include <hip/hip_bf16.h>

// B=4, S=4096, D=768, K=64
#define BB 4
#define SS 4096
#define DD 768
#define KK 64
#define ROWS (BB*SS)   // 16384
#define NSPLIT 8       // split-K factor for k_y

typedef __attribute__((ext_vector_type(16))) __bf16 v16bf;
typedef __attribute__((ext_vector_type(8)))  __bf16 v8bf;
typedef __attribute__((ext_vector_type(8)))  float  v8f;

__device__ __forceinline__ v8f wmma_bf16(v16bf a, v16bf b, v8f c) {
    return __builtin_amdgcn_wmma_f32_16x16x32_bf16(false, a, false, b, (short)0, c,
                                                   false, false);
}

// element e of a 16-bit WMMA operand -> K index within a 32-deep tile
__device__ __forceinline__ int kmap(int e, int kh) {
    return (e < 8 ? e : e + 8) + 8 * kh;
}

// A-tile (16x32 bf16), row-major source: two contiguous 16B runs -> b128 loads
__device__ __forceinline__ v16bf load_tileA(const __bf16* __restrict__ src,
                                            int ld, int m0, int k0) {
    int lane = threadIdx.x & 31;
    int m = lane & 15, kh = lane >> 4;
    const __bf16* p = src + (size_t)(m0 + m) * ld + k0 + 8 * kh;
    v8bf lo = *(const v8bf*)p;
    v8bf hi = *(const v8bf*)(p + 16);
    v16bf a;
#pragma unroll
    for (int e = 0; e < 8; ++e) { a[e] = lo[e]; a[8 + e] = hi[e]; }
    return a;
}

// B-tile from pre-swizzled storage: lane L reads its 16 elements contiguously.
// packed layout: tile (kt,nt) at ((kt*Ntiles + nt)*512), element (lane,e) at lane*16+e
__device__ __forceinline__ v16bf load_tileB_packed(const __bf16* __restrict__ base,
                                                   int Ntiles, int kt, int nt) {
    const __bf16* p = base + (((size_t)kt * Ntiles + nt) << 9) + (threadIdx.x & 31) * 16;
    return *(const v16bf*)p;
}

// scatter address of element (k within [0,Kdim), n) in packed-B storage
__device__ __forceinline__ size_t packB_addr(int k, int n, int Ntiles) {
    int kt = k >> 5, kl = k & 31;
    int khc = (kl >> 3) & 1;
    int e = (kl & 7) + ((kl >> 4) & 1) * 8;
    int lanec = (n & 15) + (khc << 4);
    return (((size_t)kt * Ntiles + (n >> 4)) << 9) + lanec * 16 + e;
}

// ---------- elementwise f32 -> bf16 (n multiple of 2048) ----------
__global__ void k_cvt(const float* __restrict__ in, __bf16* __restrict__ out) {
    int i = (blockIdx.x * blockDim.x + threadIdx.x) * 8;
    float4 a = *(const float4*)(in + i);
    float4 b = *(const float4*)(in + i + 4);
    v8bf o;
    o[0] = (__bf16)a.x; o[1] = (__bf16)a.y; o[2] = (__bf16)a.z; o[3] = (__bf16)a.w;
    o[4] = (__bf16)b.x; o[5] = (__bf16)b.y; o[6] = (__bf16)b.z; o[7] = (__bf16)b.w;
    *(v8bf*)(out + i) = o;
}

// ---------- pack f32 [Kdim x N] row-major into WMMA-B swizzled bf16 tiles ----------
__global__ void k_pack_b(const float* __restrict__ src, __bf16* __restrict__ dst, int N) {
    int i = blockIdx.x * 256 + threadIdx.x;      // element in packed order
    int tile = i >> 9, r = i & 511;
    int lane = r >> 4, e = r & 15;
    int Ntiles = N >> 4;
    int kt = tile / Ntiles, nt = tile - kt * Ntiles;
    int k = kt * 32 + kmap(e, lane >> 4);
    dst[i] = (__bf16)src[(size_t)k * N + nt * 16 + (lane & 15)];
}

// ---------- row sums of squares (f32 inputs, exact) ----------
__global__ void k_sumsq(const float* __restrict__ x, const float* __restrict__ pos,
                        float* __restrict__ x2, float* __restrict__ p2) {
    int i = blockIdx.x * blockDim.x + threadIdx.x;
    const float* src; float* dst; int idx;
    if (i < ROWS)            { src = x   + (size_t)i * DD;          dst = x2; idx = i; }
    else if (i < ROWS + KK)  { src = pos + (size_t)(i - ROWS) * DD; dst = p2; idx = i - ROWS; }
    else return;
    float s = 0.f;
    for (int d = 0; d < DD; d += 4) {
        float4 v = *(const float4*)(src + d);
        s += v.x * v.x + v.y * v.y + v.z * v.z + v.w * v.w;
    }
    dst[idx] = s;
}

// ---------- d = sqrt(max(x2 + p2 - 2*x.p, 1e-12))  [ROWS x 64] ----------
__global__ void k_dist(const __bf16* __restrict__ xbf, const __bf16* __restrict__ posbf,
                       const float* __restrict__ x2, const float* __restrict__ p2,
                       float* __restrict__ dists) {
    int wave = threadIdx.x >> 5;
    int row0 = blockIdx.x * 128 + wave * 16;
    v8f acc[4] = {};
    for (int kk = 0; kk < DD; kk += 32) {
        v16bf a = load_tileA(xbf, DD, row0, kk);
#pragma unroll
        for (int nt = 0; nt < 4; ++nt) {
            v16bf b = load_tileA(posbf, DD, nt * 16, kk);  // B = pos^T: contiguous
            acc[nt] = wmma_bf16(a, b, acc[nt]);
        }
    }
    int lane = threadIdx.x & 31, n = lane & 15, kh = lane >> 4;
#pragma unroll
    for (int nt = 0; nt < 4; ++nt)
#pragma unroll
        for (int r = 0; r < 8; ++r) {
            int row = row0 + r + 8 * kh;
            int col = nt * 16 + n;
            float d2 = x2[row] + p2[col] - 2.0f * acc[nt][r];
            dists[(size_t)row * KK + col] = sqrtf(fmaxf(d2, 1e-12f));
        }
}

// ---------- per-splat 0.2-quantile via exact 3-pass radix-histogram select ----------
__global__ void k_radius(const float* __restrict__ dists, float* __restrict__ radius) {
    __shared__ unsigned int hist[2048];
    __shared__ unsigned int sel_bucket, sel_below;
    const int k = blockIdx.x;
    const int t = threadIdx.x;
    float vals[2];
    const int      shifts[3]  = {21, 10, 0};
    const unsigned nbucket[3] = {2048u, 2048u, 1024u};
    for (int ri = 0; ri < 2; ++ri) {
        unsigned rank = 3276u + (unsigned)ri;
        unsigned prefix = 0, himask = 0;
        for (int pass = 0; pass < 3; ++pass) {
            int shift = shifts[pass];
            unsigned nb = nbucket[pass];
            for (int i = t; i < 2048; i += 256) hist[i] = 0u;
            __syncthreads();
            for (int i = t; i < ROWS; i += 256) {
                unsigned v = __float_as_uint(dists[(size_t)i * KK + k]);
                if ((v & himask) == prefix)
                    atomicAdd(&hist[(v >> shift) & (nb - 1u)], 1u);
            }
            __syncthreads();
            if (t == 0) {
                unsigned cum = 0, b = 0;
                for (; b < nb; ++b) {
                    unsigned c = hist[b];
                    if (rank < cum + c) break;
                    cum += c;
                }
                sel_bucket = b; sel_below = cum;
            }
            __syncthreads();
            prefix |= (sel_bucket << shift);
            rank   -= sel_below;
            himask |= ((nb - 1u) << shift);
            __syncthreads();
        }
        vals[ri] = __uint_as_float(prefix);
    }
    if (t == 0) {
        float r = vals[0] + 0.6f * (vals[1] - vals[0]);
        radius[k] = fminf(fmaxf(r, 0.5f), 8.0f);
    }
}

// ---------- w in both layouts, stored bf16 (consistent num/denom) ----------
__global__ void k_weights(const float* __restrict__ dists, const float* __restrict__ radius,
                          __bf16* __restrict__ wbf, __bf16* __restrict__ wT) {
    __shared__ float rad[KK];
    int t = threadIdx.x;
    if (t < KK) rad[t] = radius[t];
    __syncthreads();
    int row = blockIdx.x * 256 + t;          // [0, ROWS)
    int b = row >> 12, s = row & (SS - 1);
    for (int k = 0; k < KK; ++k) {
        float d = dists[(size_t)row * KK + k];
        float r = rad[k];
        float wv = (d < r) ? expf(-d / (0.3f * r)) : 0.f;
        __bf16 wb = (__bf16)wv;
        wbf[(size_t)row * KK + k] = wb;
        wT[((size_t)(b * KK + k) << 12) + s] = wb;
    }
}

// ---------- colsum[b][k] = sum_s w (deterministic tree reduction) ----------
__global__ void k_colsum(const __bf16* __restrict__ wT, float* __restrict__ colsum) {
    __shared__ float red[256];
    int b = blockIdx.y, k = blockIdx.x, t = threadIdx.x;
    const __bf16* src = wT + ((size_t)(b * KK + k)) * SS;
    float s = 0.f;
    for (int i = t; i < SS; i += 256) s += (float)src[i];
    red[t] = s; __syncthreads();
    for (int off = 128; off > 0; off >>= 1) {
        if (t < off) red[t] += red[t + off];
        __syncthreads();
    }
    if (t == 0) colsum[b * KK + k] = red[0];
}

// ---------- v = x @ Wv, scattered directly into packed-B layout (per batch) ----------
__global__ void k_gemm_xwv(const __bf16* __restrict__ xbf, const __bf16* __restrict__ Wvp,
                           __bf16* __restrict__ vpack) {
    int wave = threadIdx.x >> 5;
    int row0 = blockIdx.x * 128 + wave * 16;
    int col0 = blockIdx.y * 128;
    v8f acc[8] = {};
    for (int kk = 0; kk < DD; kk += 32) {
        v16bf a = load_tileA(xbf, DD, row0, kk);
#pragma unroll
        for (int nt = 0; nt < 8; ++nt) {
            v16bf bt = load_tileB_packed(Wvp, DD / 16, kk >> 5, (col0 >> 4) + nt);
            acc[nt] = wmma_bf16(a, bt, acc[nt]);
        }
    }
    int lane = threadIdx.x & 31, n = lane & 15, kh = lane >> 4;
#pragma unroll
    for (int nt = 0; nt < 8; ++nt)
#pragma unroll
        for (int r = 0; r < 8; ++r) {
            int row = row0 + r + 8 * kh;
            int b = row >> 12, sb = row & (SS - 1);
            int d = col0 + nt * 16 + n;
            __bf16* vb = vpack + (size_t)b * SS * DD;
            vb[packB_addr(sb, d, DD / 16)] = (__bf16)acc[nt][r];
        }
}

// ---------- Ypart[ks][b] = w[b][s-slice]^T @ v[b][s-slice]  (split-K over S) ----------
__global__ void k_y(const __bf16* __restrict__ wT, const __bf16* __restrict__ vpack,
                    float* __restrict__ Ypart) {
    int b = blockIdx.y, ks = blockIdx.z;
    int col0 = blockIdx.x * 128;
    int wave = threadIdx.x >> 5;     // 128 threads -> 4 waves, one m-tile each
    int m0 = wave * 16;
    const __bf16* wTb = wT + (size_t)b * KK * SS;
    const __bf16* vb  = vpack + (size_t)b * SS * DD;
    int s0 = ks * (SS / NSPLIT);
    v8f acc[8] = {};
    for (int kk = 0; kk < SS / NSPLIT; kk += 32) {
        v16bf a = load_tileA(wTb, SS, m0, s0 + kk);       // contiguous along s
        int kt = (s0 + kk) >> 5;
#pragma unroll
        for (int nt = 0; nt < 8; ++nt) {
            v16bf bt = load_tileB_packed(vb, DD / 16, kt, (col0 >> 4) + nt);
            acc[nt] = wmma_bf16(a, bt, acc[nt]);
        }
    }
    int lane = threadIdx.x & 31, n = lane & 15, kh = lane >> 4;
    float* Yp = Ypart + ((size_t)ks * BB + b) * KK * DD;
#pragma unroll
    for (int nt = 0; nt < 8; ++nt)
#pragma unroll
        for (int r = 0; r < 8; ++r) {
            int m = m0 + r + 8 * kh;
            Yp[(size_t)m * DD + col0 + nt * 16 + n] = acc[nt][r];
        }
}

// ---------- Z = amp * sum_ks Ypart, written in packed-B layout ----------
__global__ void k_zred(const float* __restrict__ Ypart, const float* __restrict__ amp,
                       __bf16* __restrict__ Zpack) {
    int i = blockIdx.x * 256 + threadIdx.x;      // over BB*KK*DD
    int b = i / (KK * DD);
    int rem = i - b * KK * DD;
    int k = rem / DD, d = rem - k * DD;
    float s = 0.f;
#pragma unroll
    for (int ks = 0; ks < NSPLIT; ++ks)
        s += Ypart[((size_t)ks * BB + b) * KK * DD + rem];
    __bf16* Zb = Zpack + (size_t)b * KK * DD;
    Zb[packB_addr(k, d, DD / 16)] = (__bf16)(amp[k] * s);
}

// ---------- out = (w @ Z) / (w·(amp⊙colsum) + 1e-8) ----------
__global__ void k_out(const __bf16* __restrict__ wbf, const __bf16* __restrict__ Zpack,
                      const float* __restrict__ amp, const float* __restrict__ colsum,
                      float* __restrict__ out) {
    __shared__ float ampcol[KK];
    __shared__ float rsum[128];
    int b = blockIdx.z;
    int row0 = blockIdx.x * 128;     // row within batch
    int col0 = blockIdx.y * 128;
    int t = threadIdx.x;
    if (t < KK) ampcol[t] = amp[t] * colsum[b * KK + t];
    __syncthreads();
    const __bf16* wb = wbf + (size_t)b * SS * KK;
    if (t < 128) {
        const __bf16* wr = wb + (size_t)(row0 + t) * KK;
        float s = 0.f;
        for (int k = 0; k < KK; ++k) s += (float)wr[k] * ampcol[k];
        rsum[t] = s + 1e-8f;
    }
    __syncthreads();
    int wave = t >> 5;
    int wrow0 = row0 + wave * 16;
    const __bf16* Zb = Zpack + (size_t)b * KK * DD;
    v8f acc[8] = {};
#pragma unroll
    for (int kk = 0; kk < KK; kk += 32) {
        v16bf a = load_tileA(wb, KK, wrow0, kk);
#pragma unroll
        for (int nt = 0; nt < 8; ++nt) {
            v16bf bt = load_tileB_packed(Zb, DD / 16, kk >> 5, (col0 >> 4) + nt);
            acc[nt] = wmma_bf16(a, bt, acc[nt]);
        }
    }
    int lane = t & 31, n = lane & 15, kh = lane >> 4;
    float* ob = out + (size_t)b * SS * DD;
#pragma unroll
    for (int nt = 0; nt < 8; ++nt)
#pragma unroll
        for (int r = 0; r < 8; ++r) {
            int row = wrow0 + r + 8 * kh;
            ob[(size_t)row * DD + col0 + nt * 16 + n] =
                acc[nt][r] / rsum[row - row0];
        }
}

extern "C" void kernel_launch(void* const* d_in, const int* in_sizes, int n_in,
                              void* d_out, int out_size, void* d_ws, size_t ws_size,
                              hipStream_t stream) {
    (void)in_sizes; (void)n_in; (void)out_size; (void)ws_size;
    const float* x   = (const float*)d_in[0];   // [4,4096,768]
    const float* pos = (const float*)d_in[1];   // [64,768]
    const float* amp = (const float*)d_in[2];   // [64]
    const float* Wv  = (const float*)d_in[3];   // [768,768]
    float* out = (float*)d_out;                 // [4,4096,768]

    // ---- workspace layout (bytes), all sub-buffers >=256B aligned ----
    char* p = (char*)d_ws;
    float* dists  = (float*)p;  p += (size_t)ROWS * KK * 4;   // 4 MB
    float* x2     = (float*)p;  p += (size_t)ROWS * 4;
    float* p2     = (float*)p;  p += KK * 4;
    float* radius = (float*)p;  p += KK * 4;
    float* colsum = (float*)p;  p += BB * KK * 4;
    __bf16* xbf   = (__bf16*)p; p += (size_t)ROWS * DD * 2;   // 24 MB
    __bf16* posbf = (__bf16*)p; p += (size_t)KK * DD * 2;
    __bf16* Wvp   = (__bf16*)p; p += (size_t)DD * DD * 2;     // packed-B
    __bf16* wbf   = (__bf16*)p; p += (size_t)ROWS * KK * 2;   // 2 MB
    __bf16* wT    = (__bf16*)p; p += (size_t)ROWS * KK * 2;   // 2 MB
    __bf16* vpack = (__bf16*)p; p += (size_t)ROWS * DD * 2;   // 24 MB packed-B per batch
    __bf16* Zpack = (__bf16*)p; p += (size_t)BB * KK * DD * 2;
    float* Ypart  = (float*)p;  p += (size_t)NSPLIT * BB * KK * DD * 4;  // 6 MB

    k_cvt   <<<(ROWS * DD) / (8 * 256), 256, 0, stream>>>(x,   xbf);
    k_cvt   <<<(KK * DD)   / (8 * 256), 256, 0, stream>>>(pos, posbf);
    k_pack_b<<<(DD * DD) / 256, 256, 0, stream>>>(Wv, Wvp, DD);

    k_sumsq   <<<(ROWS + KK + 255) / 256, 256, 0, stream>>>(x, pos, x2, p2);
    k_dist    <<<ROWS / 128, 256, 0, stream>>>(xbf, posbf, x2, p2, dists);
    k_radius  <<<KK, 256, 0, stream>>>(dists, radius);
    k_weights <<<ROWS / 256, 256, 0, stream>>>(dists, radius, wbf, wT);
    k_colsum  <<<dim3(KK, BB), 256, 0, stream>>>(wT, colsum);
    k_gemm_xwv<<<dim3(ROWS / 128, DD / 128), 256, 0, stream>>>(xbf, Wvp, vpack);
    k_y       <<<dim3(DD / 128, BB, NSPLIT), 128, 0, stream>>>(wT, vpack, Ypart);
    k_zred    <<<(BB * KK * DD) / 256, 256, 0, stream>>>(Ypart, amp, Zpack);
    k_out     <<<dim3(SS / 128, DD / 128, BB), 256, 0, stream>>>(wbf, Zpack, amp, colsum, out);
}
